// GaussianBasis_37615323579057
// MI455X (gfx1250) — matrix operand order repaired and checked by reading, not compiled, not run on hardware
//
#include <hip/hip_runtime.h>
#include <hip/hip_bf16.h>
#include <stdint.h>
#include <stddef.h>

// ---------------------------------------------------------------------------
// GaussianBasis for MI455X (gfx1250, wave32, WMMA)
//   d_matrix (10000x10000) = [alpha*w]  x  psi^T   via v_wmma_f32_16x16x32_f16
//   image raster: per-gaussian 32x32 window, fp32 global atomics
// ---------------------------------------------------------------------------

typedef __attribute__((ext_vector_type(16))) _Float16 v16h;
typedef __attribute__((ext_vector_type(8)))  _Float16 v8h;
typedef __attribute__((ext_vector_type(8)))  float    v8f;

#define GN      10000
#define GC      8
#define GJ      16
#define GK      128          // C*J, GEMM K dimension
#define IMG_H   512
#define IMG_W   512
#define TWIN    32
#define NTILES  625          // 10000 / 16, exact

// ---------------------------------------------------------------------------
// Kernel 0: zero the accumulated image region of d_out
// ---------------------------------------------------------------------------
__global__ void gb_zero_img(float* __restrict__ img) {
  int i = blockIdx.x * blockDim.x + threadIdx.x;
  if (i < 3 * IMG_H * IMG_W) img[i] = 0.0f;
}

// ---------------------------------------------------------------------------
// Kernel 1: per-gaussian prep.
//  - softmax(alpha_logits)
//  - soft_color = alpha @ cluster_means
//  - detail     = einsum('nc,ncl,cld->nd')      (written directly to d_out)
//  - aw16[n,k]  = f16( alpha[n,c] * w_proj[n,c,l] ),  k = c*16+l   (GEMM A)
//  - psi16[m,k] = f16( psi[m,c,l] )                                (GEMM B)
//  - projection -> px,py,conic + color + opacity into gparams
// ---------------------------------------------------------------------------
__global__ void gb_prep(const float* __restrict__ alpha_logits,
                        const float* __restrict__ w_proj,
                        const float* __restrict__ psi,
                        const float* __restrict__ xyz,
                        const float* __restrict__ chol,
                        const float* __restrict__ opacity,
                        const float* __restrict__ cmeans,
                        const float* __restrict__ cbases,
                        _Float16* __restrict__ aw16,
                        _Float16* __restrict__ psi16,
                        float* __restrict__ gparams,
                        float* __restrict__ detail_out) {
  int n = blockIdx.x * blockDim.x + threadIdx.x;
  if (n >= GN) return;

  // --- softmax over C=8 ---
  float lg[GC], al[GC];
  float mx = -1e30f;
  #pragma unroll
  for (int c = 0; c < GC; ++c) { lg[c] = alpha_logits[n * GC + c]; mx = fmaxf(mx, lg[c]); }
  float s = 0.0f;
  #pragma unroll
  for (int c = 0; c < GC; ++c) { al[c] = expf(lg[c] - mx); s += al[c]; }
  float inv = 1.0f / s;
  #pragma unroll
  for (int c = 0; c < GC; ++c) al[c] *= inv;

  // --- soft color, detail, f16 A operand ---
  float col0 = 0.f, col1 = 0.f, col2 = 0.f, d0 = 0.f, d1 = 0.f;
  #pragma unroll
  for (int c = 0; c < GC; ++c) {
    col0 += al[c] * cmeans[c * 3 + 0];
    col1 += al[c] * cmeans[c * 3 + 1];
    col2 += al[c] * cmeans[c * 3 + 2];
    const float* wv = w_proj + ((size_t)n * GC + c) * GJ;
    const float* cb = cbases + (size_t)c * GJ * 2;
    float t0 = 0.f, t1 = 0.f;
    #pragma unroll
    for (int l = 0; l < GJ; ++l) {
      float w = wv[l];
      t0 += w * cb[l * 2 + 0];
      t1 += w * cb[l * 2 + 1];
      aw16[(size_t)n * GK + c * GJ + l] = (_Float16)(al[c] * w);
    }
    d0 += al[c] * t0;
    d1 += al[c] * t1;
  }
  detail_out[n * 2 + 0] = d0;
  detail_out[n * 2 + 1] = d1;

  // --- f16 B operand (psi flattened [m, c*16+l] is already row-major K) ---
  const float* prow = psi + (size_t)n * GK;
  #pragma unroll 4
  for (int k = 0; k < GK; ++k) psi16[(size_t)n * GK + k] = (_Float16)prow[k];

  // --- projection (project_gaussians_2d) ---
  float px = 0.5f * IMG_W * (xyz[n * 2 + 0] + 1.0f);
  float py = 0.5f * IMG_H * (xyz[n * 2 + 1] + 1.0f);
  float l0 = chol[n * 3 + 0] + 0.5f;
  float l1 = chol[n * 3 + 1];
  float l2 = chol[n * 3 + 2] + 0.5f;
  float c00 = l0 * l0, c01 = l0 * l1, c11 = l1 * l1 + l2 * l2;
  float idet = 1.0f / (c00 * c11 - c01 * c01);

  float* gp = gparams + (size_t)n * 12;
  gp[0] = px; gp[1] = py;
  gp[2] = c11 * idet; gp[3] = -c01 * idet; gp[4] = c00 * idet;
  gp[5] = col0; gp[6] = col1; gp[7] = col2;
  gp[8] = opacity[n];
}

// ---------------------------------------------------------------------------
// Kernel 2: D = A x B^T via WMMA. One wave32 per 16x16 output tile.
//  A fragment layout (16-bit A 16x32, CDNA5 ISA):
//    lane L (h=L/16, m=L%16): a[0..7]  = A[m][kc + 8h + 0..7]
//                             a[8..15] = A[m][kc + 16 + 8h + 0..7]
//  B fragment layout (K x 16, rows striped across lanes):
//    lane L (h=L/16, ncol=L%16): b[0..7] = B[kc + 8h + 0..7][ncol], b[8..15] = B[kc + 16 + 8h + 0..7][ncol]
//                              with B[k][ncol] = psi16[row = tn*16+ncol][k]
//  D layout: lane L = column (L%16); acc[r] = D[row = r + 8*(L/16)][col]
// ---------------------------------------------------------------------------
__global__ void __launch_bounds__(32) gb_gemm(const _Float16* __restrict__ aw16,
                                              const _Float16* __restrict__ psi16,
                                              float* __restrict__ dmat) {
  const int tm   = blockIdx.y;
  const int tn   = blockIdx.x;
  const int lane = threadIdx.x & 31;
  const int half = lane >> 4;
  const int r    = lane & 15;

  const _Float16* arow = aw16  + (size_t)(tm * 16 + r) * GK;
  const _Float16* brow = psi16 + (size_t)(tn * 16 + r) * GK;

  v8f acc = {};
  #pragma unroll
  for (int kc = 0; kc < GK; kc += 32) {
    // 16-byte aligned b128 loads per the fragment layouts
    v8h alo = *(const v8h*)(arow + kc + 8 * half);
    v8h ahi = *(const v8h*)(arow + kc + 16 + 8 * half);
    v8h blo = *(const v8h*)(brow + kc + 8 * half);
    v8h bhi = *(const v8h*)(brow + kc + 16 + 8 * half);
    v16h a, b;
    #pragma unroll
    for (int i = 0; i < 8; ++i) {
      a[i] = alo[i]; a[8 + i] = ahi[i];
      b[i] = blo[i]; b[8 + i] = bhi[i];
    }
    // (neg_a, A, neg_b, B, c_mod, C, reuse_a, reuse_b)
    acc = __builtin_amdgcn_wmma_f32_16x16x32_f16(false, a, false, b,
                                                 (short)0, acc, false, false);
  }

  const int col = tn * 16 + (lane & 15);
  #pragma unroll
  for (int rr = 0; rr < 8; ++rr) {
    const int row = tm * 16 + rr + 8 * half;
    dmat[(size_t)row * GN + col] = acc[rr];
  }
}

// ---------------------------------------------------------------------------
// Kernel 3: rasterize. grid = (4, N), block = 256 -> 32x32 window/gaussian.
// ---------------------------------------------------------------------------
__global__ void gb_raster(const float* __restrict__ gparams,
                          float* __restrict__ img) {
  const int n = blockIdx.y;
  const int p = blockIdx.x * blockDim.x + threadIdx.x;   // 0..1023

  const float* gp = gparams + (size_t)n * 12;
  const float px = gp[0], py = gp[1];
  const float c0 = gp[2], c1 = gp[3], c2 = gp[4];
  const float cr = gp[5], cg = gp[6], cb = gp[7];
  const float op = gp[8];

  const int tx = p & 31;
  const int ty = p >> 5;
  const int xi = (int)floorf(px) + tx - TWIN / 2;
  const int yi = (int)floorf(py) + ty - TWIN / 2;

  const float dx = px - ((float)xi + 0.5f);
  const float dy = py - ((float)yi + 0.5f);
  const float sigma = 0.5f * (c0 * dx * dx + c2 * dy * dy) + c1 * dx * dy;
  const float a = fminf(0.999f, op * expf(-sigma));

  const bool valid = (sigma >= 0.0f) && (a > (1.0f / 255.0f)) &&
                     (yi >= 0) && (yi < IMG_H) && (xi >= 0) && (xi < IMG_W);
  if (valid) {
    const size_t pix = (size_t)yi * IMG_W + xi;
    atomicAdd(img + 0 * (size_t)IMG_H * IMG_W + pix, a * cr);
    atomicAdd(img + 1 * (size_t)IMG_H * IMG_W + pix, a * cg);
    atomicAdd(img + 2 * (size_t)IMG_H * IMG_W + pix, a * cb);
  }
}

// ---------------------------------------------------------------------------
extern "C" void kernel_launch(void* const* d_in, const int* in_sizes, int n_in,
                              void* d_out, int out_size, void* d_ws, size_t ws_size,
                              hipStream_t stream) {
  (void)in_sizes; (void)n_in; (void)out_size;

  const float* alpha_logits = (const float*)d_in[0];
  const float* w_proj       = (const float*)d_in[1];
  const float* psi          = (const float*)d_in[2];
  const float* xyz          = (const float*)d_in[3];
  const float* chol         = (const float*)d_in[4];
  const float* opac         = (const float*)d_in[5];
  const float* cmeans       = (const float*)d_in[6];
  const float* cbases       = (const float*)d_in[7];

  // Outputs concatenated in return order: img [3,512,512], d_matrix [N,N], detail [N,2]
  float* out    = (float*)d_out;
  float* img    = out;
  float* dmat   = out + (size_t)3 * IMG_H * IMG_W;
  float* detail = dmat + (size_t)GN * GN;

  // Workspace layout (all 256B aligned): aw16 | psi16 | gparams  (~5.6 MB)
  const size_t SZ_AW = (size_t)GN * GK * sizeof(_Float16);   // 2,560,000
  char* ws = (char*)d_ws;
  _Float16* aw16    = (_Float16*)(ws);
  _Float16* psi16   = (_Float16*)(ws + SZ_AW);
  float*    gparams = (float*)(ws + 2 * SZ_AW);
  if (ws_size < 2 * SZ_AW + (size_t)GN * 12 * sizeof(float)) return;

  // 0) zero image accumulator
  gb_zero_img<<<dim3((3 * IMG_H * IMG_W + 255) / 256), 256, 0, stream>>>(img);

  // 1) per-gaussian prep (+detail output, f16 GEMM operands)
  gb_prep<<<dim3((GN + 255) / 256), 256, 0, stream>>>(
      alpha_logits, w_proj, psi, xyz, chol, opac, cmeans, cbases,
      aw16, psi16, gparams, detail);

  // 2) WMMA GEMM: 625x625 tiles of 16x16, one wave32 each
  gb_gemm<<<dim3(NTILES, NTILES), 32, 0, stream>>>(aw16, psi16, dmat);

  // 3) raster with fp32 atomics
  gb_raster<<<dim3((TWIN * TWIN) / 256, GN), 256, 0, stream>>>(gparams, img);
}
